// RTDETRTransformerv2_89885075571165
// MI455X (gfx1250) — matrix operand order repaired and checked by reading, not compiled, not run on hardware
//
#include <hip/hip_runtime.h>

typedef __attribute__((ext_vector_type(16))) _Float16 v16h;
typedef __attribute__((ext_vector_type(8)))  _Float16 v8h;
typedef __attribute__((ext_vector_type(8)))  float    v8f;

#define BATCH  16
#define LQ     500
#define DMODEL 256     // K for every GEMM in this model
#define LV     8400    // 80*80 + 40*40 + 20*20
#define NHEAD  8
#define HDIM   32
#define NLVL   3
#define NPTS   4

#define BK     32      // K step per WMMA
#define TM     64      // block tile M (4 strips of 16)
#define TN     32      // block tile N (2 strips of 16)
#define KSTEPS (DMODEL / BK)   // 8

// ---------------------------------------------------------------------------
// C = A[Mx256] * B[256xN] + bias[N]; f32 in/out, f16 WMMA, f32 accumulate.
// N is a template parameter so B's row stride folds into load immediates.
// 128 threads = 4 waves. Wave w: nstrip = w&1, mstrips {2*(w>>1), 2*(w>>1)+1}
// -> each wave reuses its B fragment across 2 WMMAs per K step.
// Double-buffered fragment-ready LDS: per 16-row/col strip, lane l owns 16
// contiguous halves at [strip*512 + l*16]; fragment loads are v16h
// (2x ds_load_b128), cooperative stores are single ds_store_b128.
//
// A fragment (ISA 16-bit A 16x32): lane l: m = l&15, elems 0..7 -> K=kb..kb+7,
//   elems 8..15 -> K = 16+kb.., kb = 8*(l>>4).
// B fragment (same K order as A): lane l: n = l&15, elems 0..7 -> K=kb..kb+7,
//   elems 8..15 -> K = 16+kb.., kb = 8*(l>>4).
// ---------------------------------------------------------------------------
template <int N>
__global__ __launch_bounds__(128) void wmma_gemm_bias(
    const float* __restrict__ A, const float* __restrict__ Bm,
    const float* __restrict__ bias, float* __restrict__ C, int M)
{
  __shared__ _Float16 Afrag[2][4 * 512];   // double-buffered, 4 m-strips
  __shared__ _Float16 Bfrag[2][2 * 512];   // double-buffered, 2 n-strips

  const int tid    = threadIdx.x;
  const int lane   = tid & 31;
  const int wave   = tid >> 5;      // 0..3
  const int nstrip = wave & 1;      // 0..1
  const int mpair  = wave >> 1;     // 0..1 -> m strips 2*mpair, 2*mpair+1

  const int brow = blockIdx.y * TM;
  const int bcol = blockIdx.x * TN;

  // B cooperative mapping: n = tid&31, kk0 = (tid>>5)*8 (8 consecutive K)
  const int bn_  = tid & 31;
  const int bkk0 = (tid >> 5) << 3;

  // Load tiles for K offset k0 into buffer bsel.
  auto load_tiles = [&](int k0, int bsel) {
    // ---- A tile 64x32 (2 chunks of 8 halves per thread)
    #pragma unroll
    for (int c = 0; c < 2; ++c) {
      const int g   = tid + c * 128;      // 0..255
      const int m   = g >> 2;             // 0..63
      const int kk0 = (g & 3) << 3;       // 0,8,16,24
      const float* src = A + (size_t)(brow + m) * DMODEL + (k0 + kk0);
      const float4 f0 = *(const float4*)(src);
      const float4 f1 = *(const float4*)(src + 4);
      v8h hv;
      hv[0] = (_Float16)f0.x; hv[1] = (_Float16)f0.y;
      hv[2] = (_Float16)f0.z; hv[3] = (_Float16)f0.w;
      hv[4] = (_Float16)f1.x; hv[5] = (_Float16)f1.y;
      hv[6] = (_Float16)f1.z; hv[7] = (_Float16)f1.w;
      const int s    = m >> 4;
      const int l    = (m & 15) + ((kk0 & 8) ? 16 : 0);
      const int off8 = (kk0 & 16) ? 8 : 0;
      *(v8h*)&Afrag[bsel][s * 512 + l * 16 + off8] = hv;
    }
    // ---- B tile 32x32: 8 loads at compile-time offsets j*N (one clause)
    {
      const float* src = Bm + (size_t)(k0 + bkk0) * N + (bcol + bn_);
      v8h hv;
      #pragma unroll
      for (int j = 0; j < 8; ++j)
        hv[j] = (_Float16)src[j * N];     // j*N is a constant immediate
      const int scol = bn_ >> 4;
      const int l    = (bn_ & 15) + ((bkk0 & 8) ? 16 : 0);
      const int off8 = (bkk0 & 16) ? 8 : 0;
      *(v8h*)&Bfrag[bsel][scol * 512 + l * 16 + off8] = hv;
    }
    // prefetch two tiles ahead -> global_prefetch_b8
    if (k0 + 2 * BK < DMODEL) {
      __builtin_prefetch(A + (size_t)(brow + (tid >> 2)) * DMODEL +
                             (k0 + 2 * BK + ((tid & 3) << 3)), 0, 1);
      __builtin_prefetch(Bm + (size_t)(k0 + 2 * BK + bkk0) * N + (bcol + bn_), 0, 1);
    }
  };

  v8f acc0 = {}, acc1 = {};

  load_tiles(0, 0);                       // prologue

  #pragma unroll
  for (int it = 0; it < KSTEPS; ++it) {
    __syncthreads();                      // buf[it&1] ready; buf[(it+1)&1] reads
                                          // from it-1 are done before overwrite
    if (it + 1 < KSTEPS)
      load_tiles((it + 1) * BK, (it + 1) & 1);   // overlap with WMMA below

    const v16h* Af = (const v16h*)Afrag[it & 1];
    const v16h* Bf = (const v16h*)Bfrag[it & 1];
    const v16h a0 = Af[(2 * mpair)     * 32 + lane];
    const v16h a1 = Af[(2 * mpair + 1) * 32 + lane];
    const v16h b  = Bf[nstrip * 32 + lane];

    acc0 = __builtin_amdgcn_wmma_f32_16x16x32_f16(
        false, a0, false, b, (short)0, acc0, false, false);
    acc1 = __builtin_amdgcn_wmma_f32_16x16x32_f16(
        false, a1, false, b, (short)0, acc1, false, false);
  }

  // C layout: VGPR r -> M = r + 8*(lane>=16); N = lane&15.
  const int cn = bcol + nstrip * 16 + (lane & 15);
  const float bv = bias ? bias[cn] : 0.0f;
  const int r0 = brow + (2 * mpair) * 16 + ((lane >> 4) << 3);
  const int r1 = r0 + 16;
  #pragma unroll
  for (int r = 0; r < 8; ++r) {
    C[(size_t)(r0 + r) * N + cn] = acc0[r] + bv;
    C[(size_t)(r1 + r) * N + cn] = acc1[r] + bv;
  }
}

// ---------------------------------------------------------------------------
// Deformable sampling: one block per (b,q); thread = (head h = t/32, chan d).
// Softmax over 12 logits per head, bilinear-gather 3 levels x 4 points.
// ---------------------------------------------------------------------------
__global__ __launch_bounds__(256) void msda_sample(
    const float* __restrict__ v,         // [B, Lv, 256] head-major channels
    const float* __restrict__ offsets,   // [B*Lq, 192] = (h, l, p, 2)
    const float* __restrict__ aw_logits, // [B*Lq, 96]  = (h, l*4+p)
    const float* __restrict__ refp,      // [B, Lq, 4]
    float* __restrict__ out_attn)        // [B*Lq, 256]
{
  const int bq = blockIdx.x;
  const int t  = threadIdx.x;
  const int h  = t >> 5;
  const int d  = t & 31;

  const float* off = offsets   + (size_t)bq * (NHEAD * NLVL * NPTS * 2) + h * (NLVL * NPTS * 2);
  const float* awl = aw_logits + (size_t)bq * (NHEAD * NLVL * NPTS)     + h * (NLVL * NPTS);

  float w[NLVL * NPTS];
  float mx = -3.402823466e+38f;
  #pragma unroll
  for (int i = 0; i < NLVL * NPTS; ++i) { w[i] = awl[i]; mx = fmaxf(mx, w[i]); }
  float sum = 0.0f;
  #pragma unroll
  for (int i = 0; i < NLVL * NPTS; ++i) { w[i] = __expf(w[i] - mx); sum += w[i]; }
  const float inv = 1.0f / sum;

  const float* rp = refp + (size_t)bq * 4;
  const float rx = rp[0], ry = rp[1];
  const float wx2 = rp[2] * 0.5f, wy2 = rp[3] * 0.5f;

  const int b = bq / LQ;
  const float* vb = v + (size_t)b * LV * DMODEL;

  const int starts[NLVL] = {0, 6400, 8000};
  const int Hs[NLVL]     = {80, 40, 20};
  const int Ws[NLVL]     = {80, 40, 20};

  float acc = 0.0f;
  #pragma unroll
  for (int l = 0; l < NLVL; ++l) {
    const int Hl = Hs[l], Wl = Ws[l];
    const float* vl = vb + (size_t)starts[l] * DMODEL;
    #pragma unroll
    for (int p = 0; p < NPTS; ++p) {
      const float ox = off[(l * NPTS + p) * 2 + 0];
      const float oy = off[(l * NPTS + p) * 2 + 1];
      const float lx = rx + ox * (wx2 * (1.0f / NPTS));
      const float ly = ry + oy * (wy2 * (1.0f / NPTS));
      const float x  = lx * (float)Wl - 0.5f;
      const float y  = ly * (float)Hl - 0.5f;
      const float x0 = floorf(x), y0 = floorf(y);
      const float fx = x - x0,  fy = y - y0;
      const int ix0 = (int)x0, iy0 = (int)y0;

      float s00 = 0.f, s01 = 0.f, s10 = 0.f, s11 = 0.f;
      const int chan = h * HDIM + d;
      if (ix0 >= 0 && ix0 < Wl && iy0 >= 0 && iy0 < Hl)
        s00 = vl[((size_t)(iy0 * Wl + ix0)) * DMODEL + chan];
      if (ix0 + 1 >= 0 && ix0 + 1 < Wl && iy0 >= 0 && iy0 < Hl)
        s01 = vl[((size_t)(iy0 * Wl + ix0 + 1)) * DMODEL + chan];
      if (ix0 >= 0 && ix0 < Wl && iy0 + 1 >= 0 && iy0 + 1 < Hl)
        s10 = vl[((size_t)((iy0 + 1) * Wl + ix0)) * DMODEL + chan];
      if (ix0 + 1 >= 0 && ix0 + 1 < Wl && iy0 + 1 >= 0 && iy0 + 1 < Hl)
        s11 = vl[((size_t)((iy0 + 1) * Wl + ix0 + 1)) * DMODEL + chan];

      const float bil = s00 * (1.f - fx) * (1.f - fy) + s01 * fx * (1.f - fy)
                      + s10 * (1.f - fx) * fy         + s11 * fx * fy;
      acc += (w[l * NPTS + p] * inv) * bil;
    }
  }
  out_attn[(size_t)bq * DMODEL + h * HDIM + d] = acc;
}

// ---------------------------------------------------------------------------
extern "C" void kernel_launch(void* const* d_in, const int* in_sizes, int n_in,
                              void* d_out, int out_size, void* d_ws, size_t ws_size,
                              hipStream_t stream) {
  const float* query = (const float*)d_in[0];
  const float* refp  = (const float*)d_in[1];
  const float* value = (const float*)d_in[2];
  const float* Wo    = (const float*)d_in[3];
  const float* bo    = (const float*)d_in[4];
  const float* Wa    = (const float*)d_in[5];
  const float* ba    = (const float*)d_in[6];
  const float* Wv    = (const float*)d_in[7];
  const float* bv    = (const float*)d_in[8];
  const float* Wout  = (const float*)d_in[9];
  const float* bout  = (const float*)d_in[10];
  // d_in[11] = value_spatial_shapes (compile-time constants here)

  const int Mv = BATCH * LV;               // 134400 (/64 = 2100)
  const int Mq = BATCH * LQ;               // 8000   (/64 = 125)
  constexpr int NO = NHEAD * NLVL * NPTS * 2;  // 192
  constexpr int NA = NHEAD * NLVL * NPTS;      // 96

  float* ws_v    = (float*)d_ws;                       // [Mv, 256]
  float* ws_off  = ws_v   + (size_t)Mv * DMODEL;       // [Mq, 192]
  float* ws_aw   = ws_off + (size_t)Mq * NO;           // [Mq, 96]
  float* ws_attn = ws_aw  + (size_t)Mq * NA;           // [Mq, 256]

  const dim3 blk(128);

  // 1) value projection: v = value @ Wv + bv
  wmma_gemm_bias<DMODEL><<<dim3(DMODEL / TN, Mv / TM), blk, 0, stream>>>(
      value, Wv, bv, ws_v, Mv);

  // 2) sampling offsets: query @ Wo + bo
  wmma_gemm_bias<NO><<<dim3(NO / TN, Mq / TM), blk, 0, stream>>>(
      query, Wo, bo, ws_off, Mq);

  // 3) attention-weight logits: query @ Wa + ba
  wmma_gemm_bias<NA><<<dim3(NA / TN, Mq / TM), blk, 0, stream>>>(
      query, Wa, ba, ws_aw, Mq);

  // 4) softmax + bilinear sampling + weighted sum
  msda_sample<<<dim3(Mq), dim3(256), 0, stream>>>(
      ws_v, ws_off, ws_aw, refp, ws_attn);

  // 5) output projection: out = attn @ Wout + bout
  wmma_gemm_bias<DMODEL><<<dim3(DMODEL / TN, Mq / TM), blk, 0, stream>>>(
      ws_attn, Wout, bout, (float*)d_out, Mq);
}
